// GConv_19911468384628
// MI455X (gfx1250) — compile-verified, matrix operand order checked by reading, not run on hardware
//
#include <hip/hip_runtime.h>
#include <hip/hip_bf16.h>

// ---------- types for WMMA ----------
typedef __attribute__((ext_vector_type(16))) __bf16        v16bf;
typedef __attribute__((ext_vector_type(8)))  float         v8f;
typedef __attribute__((ext_vector_type(8)))  unsigned short u16x8;
typedef __attribute__((ext_vector_type(16))) unsigned short u16x16;

__device__ __forceinline__ unsigned short f32_to_bf16_rne(float f) {
    unsigned int u = __float_as_uint(f);
    unsigned int r = u + 0x7FFFu + ((u >> 16) & 1u);   // round-to-nearest-even
    return (unsigned short)(r >> 16);
}

#define HDIM 256

// ---------- degree / normalization ----------
__global__ void deg_init_kernel(float* deg, int n) {
    int i = blockIdx.x * blockDim.x + threadIdx.x;
    if (i < n) deg[i] = 1.0f;                // self-loop contributes 1
}

__global__ void deg_edges_kernel(const int* __restrict__ dst, float* deg, int e) {
    int i = blockIdx.x * blockDim.x + threadIdx.x;
    if (i < e) atomicAdd(&deg[dst[i]], 1.0f);
}

__global__ void deg_rsqrt_kernel(float* deg, int n) {
    int i = blockIdx.x * blockDim.x + threadIdx.x;
    if (i < n) deg[i] = rsqrtf(deg[i]);      // deg >= 1 always (self-loops)
}

// ---------- fp32 -> bf16 conversion ----------
__global__ void cvt_bf16_kernel(const float* __restrict__ x, unsigned short* __restrict__ xb,
                                size_t total) {
    size_t i = (size_t)blockIdx.x * blockDim.x + threadIdx.x;
    if (i < total) xb[i] = f32_to_bf16_rne(x[i]);
}

// fused ReLU + bf16 convert (between layers)
__global__ void relu_bf16_kernel(const float* __restrict__ acc, unsigned short* __restrict__ zb,
                                 size_t total) {
    size_t i = (size_t)blockIdx.x * blockDim.x + threadIdx.x;
    if (i < total) zb[i] = f32_to_bf16_rne(fmaxf(acc[i], 0.0f));
}

// ---------- pack W (fp32 row-major [256][256]) into B-fragment layout ----------
// dest index: ((ktile*256 + n)*2 + half)*16 + j
// source k  : ktile*32 + 16*(j>>3) + half*8 + (j&7)
__global__ void pack_w_kernel(const float* __restrict__ W, unsigned short* __restrict__ Bp) {
    int idx = blockIdx.x * blockDim.x + threadIdx.x;   // 8*256*32 = 65536 total
    int j  = idx & 15;
    int hl = (idx >> 4) & 1;
    int n  = (idx >> 5) & 255;
    int t  = idx >> 13;
    int k  = t * 32 + ((j >> 3) << 4) + hl * 8 + (j & 7);
    Bp[idx] = f32_to_bf16_rne(W[k * HDIM + n]);
}

// ---------- WMMA GEMM: C[M,256] = A[M,256] * B[256,256], bf16 in / f32 acc ----------
// One wave per 16(M) x 64(N) strip: A fragment loaded once per K-step and reused
// across 4 WMMAs -> 2.5 b128 loads per v_wmma (A traffic cut 4x vs 1-tile waves).
__global__ void gemm_wmma_bf16(const unsigned short* __restrict__ A,   // [M][256] bf16 row-major
                               const unsigned short* __restrict__ Bp,  // packed fragments
                               float* __restrict__ C, int stripsTotal) {
    const int wave = threadIdx.x >> 5;                 // 8 waves per block
    const int lane = threadIdx.x & 31;
    const int hl   = lane >> 4;                        // half of wave
    const int m16  = lane & 15;
    const int strip = blockIdx.x * 8 + wave;
    if (strip >= stripsTotal) return;                  // wave-uniform: EXEC stays all-ones
    const int tm = strip >> 2;                         // 4 N-strips of 64 cols each
    const int tn = strip & 3;

    const unsigned short* arow = A + (size_t)(tm * 16 + m16) * HDIM;
    // 4 B fragment bases, one per 16-col group of this strip
    const unsigned short* bbase[4];
#pragma unroll
    for (int c = 0; c < 4; ++c)
        bbase[c] = Bp + ((size_t)(tn * 64 + c * 16 + m16) * 2 + hl) * 16;

    v8f acc0 = {}, acc1 = {}, acc2 = {}, acc3 = {};
#pragma unroll
    for (int kt = 0; kt < 8; ++kt) {
        const int k0 = kt * 32;
        // A fragment (ISA 16-bit A 16x32 layout): lanes 0-15 hold K 0..7 & 16..23,
        // lanes 16-31 hold K 8..15 & 24..31 -> two aligned b128 loads per lane.
        u16x8 a_lo = *(const u16x8*)(arow + k0 + hl * 8);
        u16x8 a_hi = *(const u16x8*)(arow + k0 + 16 + hl * 8);
        u16x16 au;
#pragma unroll
        for (int i = 0; i < 8; ++i) { au[i] = a_lo[i]; au[i + 8] = a_hi[i]; }
        v16bf av = __builtin_bit_cast(v16bf, au);

        const size_t koff = (size_t)kt * (HDIM * 32);
#pragma unroll
        for (int c = 0; c < 4; ++c) {
            const unsigned short* bptr = bbase[c] + koff;
            u16x8 b_lo = *(const u16x8*)(bptr);
            u16x8 b_hi = *(const u16x8*)(bptr + 8);
            u16x16 bu;
#pragma unroll
            for (int i = 0; i < 8; ++i) { bu[i] = b_lo[i]; bu[i + 8] = b_hi[i]; }
            v16bf bv = __builtin_bit_cast(v16bf, bu);
            v8f cin = (c == 0) ? acc0 : (c == 1) ? acc1 : (c == 2) ? acc2 : acc3;
            v8f cout = __builtin_amdgcn_wmma_f32_16x16x32_bf16(
                /*neg_a=*/false, av, /*neg_b=*/false, bv,
                /*c_mod=*/(short)0, cin, /*reuse_a=*/false, /*reuse_b=*/false);
            if (c == 0) acc0 = cout; else if (c == 1) acc1 = cout;
            else if (c == 2) acc2 = cout; else acc3 = cout;
        }
    }

    // C/D layout: lane half selects M 0..7 vs 8..15; VGPR r = row, lane&15 = col.
#pragma unroll
    for (int c = 0; c < 4; ++c) {
        const int col = tn * 64 + c * 16 + m16;
        v8f accv = (c == 0) ? acc0 : (c == 1) ? acc1 : (c == 2) ? acc2 : acc3;
#pragma unroll
        for (int r = 0; r < 8; ++r)
            C[(size_t)(tm * 16 + hl * 8 + r) * HDIM + col] = accv[r];
    }
}

// ---------- aggregation init: acc = bias + h * dis^2 (self-loop term) ----------
__global__ void agg_init_kernel(const float* __restrict__ h, const float* __restrict__ dis,
                                const float* __restrict__ bias, float* __restrict__ acc) {
    size_t i = (size_t)blockIdx.x * HDIM + threadIdx.x;   // grid = N blocks of 256
    int row = blockIdx.x;
    int f   = threadIdx.x;
    float d = dis[row];
    acc[i] = bias[f] + h[i] * d * d;
}

// ---------- edge scatter: acc[dst] += h[src] * (dis[src]*dis[dst]) ----------
// One wave per edge; h row is L2-resident (51.2 MB << 192 MB L2).
__global__ void scatter_edges_kernel(const int* __restrict__ src, const int* __restrict__ dst,
                                     const float* __restrict__ dis, const float* __restrict__ h,
                                     float* __restrict__ acc, int E) {
    int wave = threadIdx.x >> 5;
    int lane = threadIdx.x & 31;
    int e = blockIdx.x * 8 + wave;
    if (e >= E) return;
    int s = src[e], d = dst[e];
    float w = dis[s] * dis[d];
    const float* hs = h  + (size_t)s * HDIM;
    float*       ad = acc + (size_t)d * HDIM;
#pragma unroll
    for (int f = lane; f < HDIM; f += 32)
        atomicAdd(&ad[f], hs[f] * w);
}

// ---------- final row-wise L2 normalize ----------
__global__ void l2norm_kernel(const float* __restrict__ acc, float* __restrict__ out) {
    int row = blockIdx.x;
    int tid = threadIdx.x;
    float v = acc[(size_t)row * HDIM + tid];
    float s = v * v;
#pragma unroll
    for (int m = 16; m > 0; m >>= 1) s += __shfl_xor(s, m, 32);
    __shared__ float ws[8];
    if ((tid & 31) == 0) ws[tid >> 5] = s;
    __syncthreads();
    float tot = 0.0f;
#pragma unroll
    for (int i = 0; i < 8; ++i) tot += ws[i];
    float inv = 1.0f / fmaxf(sqrtf(tot), 1e-12f);
    out[(size_t)row * HDIM + tid] = v * inv;
}

// ---------- host launcher ----------
extern "C" void kernel_launch(void* const* d_in, const int* in_sizes, int n_in,
                              void* d_out, int out_size, void* d_ws, size_t ws_size,
                              hipStream_t stream) {
    const float* x   = (const float*)d_in[0];
    const int*   ei  = (const int*)d_in[1];       // [2][E] flat
    const float* W1  = (const float*)d_in[2];
    const float* b1  = (const float*)d_in[3];
    const float* W2  = (const float*)d_in[4];
    const float* b2  = (const float*)d_in[5];
    float* out = (float*)d_out;

    const int N = in_sizes[0] / HDIM;
    const int E = in_sizes[1] / 2;
    const int* src = ei;
    const int* dst = ei + E;

    // carve scratch (256B aligned)
    char* p = (char*)d_ws;
    auto take = [&](size_t bytes) { void* r = p; p += (bytes + 255) & ~(size_t)255; return r; };
    unsigned short* xb  = (unsigned short*)take((size_t)N * HDIM * 2); // bf16 activations
    float*          h   = (float*)take((size_t)N * HDIM * 4);          // GEMM output
    float*          acc = (float*)take((size_t)N * HDIM * 4);          // aggregation buffer
    unsigned short* wp1 = (unsigned short*)take(8 * 256 * 32 * 2);
    unsigned short* wp2 = (unsigned short*)take(8 * 256 * 32 * 2);
    float*          dis = (float*)take((size_t)N * 4);                 // deg -> rsqrt in place

    const size_t total = (size_t)N * HDIM;
    const int tilesM = (N + 15) / 16;
    const int stripsTotal = tilesM * 4;               // 4 strips of 64 cols per M-tile
    const int gemmBlocks = (stripsTotal + 7) / 8;
    const int edgeBlocks = (E + 7) / 8;

    // normalization coefficients
    deg_init_kernel<<<(N + 255) / 256, 256, 0, stream>>>(dis, N);
    deg_edges_kernel<<<(E + 255) / 256, 256, 0, stream>>>(dst, dis, E);
    deg_rsqrt_kernel<<<(N + 255) / 256, 256, 0, stream>>>(dis, N);

    // convert + pack
    cvt_bf16_kernel<<<(int)((total + 255) / 256), 256, 0, stream>>>(x, xb, total);
    pack_w_kernel<<<256, 256, 0, stream>>>(W1, wp1);
    pack_w_kernel<<<256, 256, 0, stream>>>(W2, wp2);

    // ---- layer 1 ----
    gemm_wmma_bf16<<<gemmBlocks, 256, 0, stream>>>(xb, wp1, h, stripsTotal);
    agg_init_kernel<<<N, 256, 0, stream>>>(h, dis, b1, acc);
    scatter_edges_kernel<<<edgeBlocks, 256, 0, stream>>>(src, dst, dis, h, acc, E);
    relu_bf16_kernel<<<(int)((total + 255) / 256), 256, 0, stream>>>(acc, xb, total);

    // ---- layer 2 ----
    gemm_wmma_bf16<<<gemmBlocks, 256, 0, stream>>>(xb, wp2, h, stripsTotal);
    agg_init_kernel<<<N, 256, 0, stream>>>(h, dis, b2, acc);
    scatter_edges_kernel<<<edgeBlocks, 256, 0, stream>>>(src, dst, dis, h, acc, E);

    // ---- normalize ----
    l2norm_kernel<<<N, 256, 0, stream>>>(acc, out);
}
